// RES_EQGATModel_61383672594540
// MI455X (gfx1250) — matrix operand order repaired and checked by reading, not compiled, not run on hardware
//
#include <hip/hip_runtime.h>
#include <math.h>

// ---------------------------------------------------------------------------
// EQGAT-style GNN on MI455X (gfx1250).
// HBM-bound (~3 GB traffic vs ~150 GFLOP) -> fp16 activations/gathers,
// all GEMMs on v_wmma_f32_16x16x32_f16 with fragment-ready packed weights,
// edge MLP fused through LDS with async global->LDS gather (ASYNCcnt).
// ---------------------------------------------------------------------------

#define NN 25000
#define NE 400000
#define SD 128
#define VD 16
#define KR 32
#define RCUT 4.5f
#define FPI 3.14159265358979323846f

typedef __attribute__((ext_vector_type(16))) _Float16 v16h;
typedef __attribute__((ext_vector_type(8)))  float    v8f;

// ---- WMMA fragment helpers (layouts per CDNA5 ISA 7.12.2, wave32) ----------

// A: 16x32 f16 tile, row-major in LDS with given row stride.
// lanes 0-15: M=lane, K=[0..7|16..23]; lanes 16-31: M=lane-16, K=[8..15|24..31]
__device__ __forceinline__ v16h ldsA(const _Float16* t, int stride) {
  const int lane = threadIdx.x & 31;
  const int m    = lane & 15;
  const int kb   = (lane & 16) ? 8 : 0;
  const _Float16* p = t + m * stride;
  v16h a;
#pragma unroll
  for (int i = 0; i < 8; ++i) { a[i] = p[kb + i]; a[i + 8] = p[16 + kb + i]; }
  return a;
}

// B: packed fragment-ready layout -- lane's 16 f16 are contiguous (32B load).
__device__ __forceinline__ v16h ldBp(const _Float16* wp) {
  return *(const v16h*)(wp + (size_t)(threadIdx.x & 31) * 16);
}

__device__ __forceinline__ float siluf(float x) { return x / (1.0f + __expf(-x)); }

// ---- elementwise / setup kernels -------------------------------------------

__global__ void k_fill_f32(float* p, float v, size_t n) {
  size_t i = (size_t)blockIdx.x * blockDim.x + threadIdx.x;
  if (i < n) p[i] = v;
}
__global__ void k_fill_u32(unsigned* p, unsigned v, size_t n) {
  size_t i = (size_t)blockIdx.x * blockDim.x + threadIdx.x;
  if (i < n) p[i] = v;
}
__global__ void k_cvt_f16(const float* __restrict__ s, _Float16* __restrict__ d, size_t n) {
  size_t i = (size_t)blockIdx.x * blockDim.x + threadIdx.x;
  if (i < n) d[i] = (_Float16)s[i];
}

// f32 [din,dout] row-major -> f16 fragment-packed [kt][ct][lane][16], padded.
// value(kt,ct,lane,e) = W[kt*32 + ((lane&16)?8:0) + (e&7) + ((e&8)?16:0)][ct*16 + (lane&15)]
__global__ void k_cvt_pack(const float* __restrict__ s, _Float16* __restrict__ d,
                           int din, int dout, int din_p, int dout_p) {
  size_t i = (size_t)blockIdx.x * blockDim.x + threadIdx.x;
  if (i >= (size_t)din_p * dout_p) return;
  int ct_n  = dout_p >> 4;
  int tile  = (int)(i >> 9);          // / 512
  int within = (int)(i & 511);
  int lane = within >> 4, e = within & 15;
  int kt = tile / ct_n, ct = tile % ct_n;
  int k = kt * 32 + ((lane & 16) ? 8 : 0) + (e & 7) + ((e & 8) ? 16 : 0);
  int n = ct * 16 + (lane & 15);
  d[i] = (k < din && n < dout) ? (_Float16)s[(size_t)k * dout + n] : (_Float16)0.f;
}

__global__ void k_embed(const int* __restrict__ at, const float* __restrict__ emb,
                        float* __restrict__ s, int nN) {
  size_t i = (size_t)blockIdx.x * blockDim.x + threadIdx.x;
  if (i >= (size_t)nN * SD) return;
  int n = (int)(i / SD), c = (int)(i % SD);
  s[i] = emb[(size_t)at[n] * SD + c];
}

// ---- edge geometry: d, r, bessel*cutoff (fp16) -----------------------------

__global__ void k_edge_geom(const float* __restrict__ pos, const int* __restrict__ src,
                            const int* __restrict__ dst, float* __restrict__ d,
                            float* __restrict__ r, _Float16* __restrict__ deh, int nE) {
  int e = blockIdx.x * blockDim.x + threadIdx.x;
  if (e >= nE) return;
  int s = src[e], t = dst[e];
  float rx = pos[3 * s + 0] - pos[3 * t + 0];
  float ry = pos[3 * s + 1] - pos[3 * t + 1];
  float rz = pos[3 * s + 2] - pos[3 * t + 2];
  float dd = sqrtf(rx * rx + ry * ry + rz * rz + 1e-12f);
  d[e] = dd;
  r[3 * e + 0] = rx / dd; r[3 * e + 1] = ry / dd; r[3 * e + 2] = rz / dd;
  float rs = dd / RCUT, env = 0.f;
  if (rs < 1.f) {
    float rs2 = rs * rs, rs6 = rs2 * rs2 * rs2;
    env = 1.f - 28.f * rs6 + 48.f * rs6 * rs - 21.f * rs6 * rs2;
  }
  float coef = env * sqrtf(2.0f / RCUT) / dd;
  float w = FPI * dd / RCUT;
#pragma unroll
  for (int k = 0; k < KR; ++k)
    deh[(size_t)e * KR + k] = (_Float16)(coef * __sinf(w * (float)(k + 1)));
}

// ---- generic node GEMM: out = act(inh[n,din] @ Wp + b), packed weights -----

__global__ __launch_bounds__(128) void k_node_gemm(
    const _Float16* __restrict__ inh, const _Float16* __restrict__ Wp,
    const float* __restrict__ bias, float* __restrict__ outf,
    _Float16* __restrict__ outh, int n, int din, int dout_p, int dout_r, int act) {
  __shared__ _Float16 tile[16 * 288];
  const int r0 = blockIdx.x * 16;
  for (int idx = threadIdx.x; idx < 16 * din; idx += 128) {
    int i = idx / din, c = idx % din;
    int row = r0 + i;
    tile[idx] = (row < n) ? inh[(size_t)row * din + c] : (_Float16)0.f;
  }
  __syncthreads();
  const int wave = threadIdx.x >> 5, lane = threadIdx.x & 31;
  const int kt_n = din >> 5, ct_n = dout_p >> 4;
  for (int ct = wave; ct < ct_n; ct += 4) {
    v8f acc = {};
    for (int kt = 0; kt < kt_n; ++kt) {
      v16h a = ldsA(tile + kt * 32, din);
      v16h b = ldBp(Wp + ((size_t)kt * ct_n + ct) * 512);
      acc = __builtin_amdgcn_wmma_f32_16x16x32_f16(false, a, false, b, (short)0,
                                                   acc, false, false);
    }
    int nn = lane & 15, mb = (lane & 16) ? 8 : 0;
    int col = ct * 16 + nn;
    float bv = (bias != nullptr && col < dout_r) ? bias[col] : 0.f;
#pragma unroll
    for (int i = 0; i < 8; ++i) {
      int row = r0 + mb + i;
      if (row < n && col < dout_r) {
        float x = acc[i] + bv;
        if (act == 1) x = siluf(x);
        if (outf) outf[(size_t)row * dout_r + col] = x;
        if (outh) outh[(size_t)row * dout_r + col] = (_Float16)x;
      }
    }
  }
}

// ---- fused edge MLP: async gather(288) -> W1+silu -> W2 -> (a, vij) --------
// NE % 16 == 0, so every edge tile is full (no zero-fill needed).

__global__ __launch_bounds__(128) void k_edge_mlp(
    const _Float16* __restrict__ sh, const _Float16* __restrict__ deh,
    const int* __restrict__ src, const int* __restrict__ dst,
    const _Float16* __restrict__ W1p, const float* __restrict__ b1,
    const _Float16* __restrict__ W2p, const float* __restrict__ b2,
    float* __restrict__ abuf, float* __restrict__ vijbuf, int nE, int dout2) {
  __shared__ _Float16 ain[16 * 288];
  __shared__ _Float16 hsm[16 * 128];
  const int e0 = blockIdx.x * 16;
  // async gather: per row 36 chunks of 16B: s[dst](16) | s[src](16) | de(4)
  for (int idx = threadIdx.x; idx < 16 * 36; idx += 128) {
    int i = idx / 36, ch = idx % 36;
    int e = e0 + i;
    const _Float16* g;
    if (ch < 16)      g = sh  + (size_t)dst[e] * SD + ch * 8;
    else if (ch < 32) g = sh  + (size_t)src[e] * SD + (ch - 16) * 8;
    else              g = deh + (size_t)e * KR + (ch - 32) * 8;
    unsigned lva = (unsigned)(uintptr_t)(const void*)(ain + i * 288 + ch * 8);
    asm volatile("global_load_async_to_lds_b128 %0, %1, off"
                 :: "v"(lva), "v"(g) : "memory");
  }
  asm volatile("s_wait_asynccnt 0" ::: "memory");
  __syncthreads();
  const int wave = threadIdx.x >> 5, lane = threadIdx.x & 31;
  // GEMM1: 288 -> 128, silu, stage h in LDS as f16
  for (int ct = wave; ct < 8; ct += 4) {
    v8f acc = {};
#pragma unroll
    for (int kt = 0; kt < 9; ++kt) {
      v16h a = ldsA(ain + kt * 32, 288);
      v16h b = ldBp(W1p + ((size_t)kt * 8 + ct) * 512);
      acc = __builtin_amdgcn_wmma_f32_16x16x32_f16(false, a, false, b, (short)0,
                                                   acc, false, false);
    }
    int nn = lane & 15, mb = (lane & 16) ? 8 : 0;
    int col = ct * 16 + nn;
    float bv = b1[col];
#pragma unroll
    for (int i = 0; i < 8; ++i)
      hsm[(mb + i) * SD + col] = (_Float16)siluf(acc[i] + bv);
  }
  __syncthreads();
  // GEMM2: 128 -> dout2 (144/176); split store into a (128) and vij tail
  const int ct_n = dout2 >> 4, vw = dout2 - SD;
  for (int ct = wave; ct < ct_n; ct += 4) {
    v8f acc = {};
#pragma unroll
    for (int kt = 0; kt < 4; ++kt) {
      v16h a = ldsA(hsm + kt * 32, SD);
      v16h b = ldBp(W2p + ((size_t)kt * ct_n + ct) * 512);
      acc = __builtin_amdgcn_wmma_f32_16x16x32_f16(false, a, false, b, (short)0,
                                                   acc, false, false);
    }
    int nn = lane & 15, mb = (lane & 16) ? 8 : 0;
    int col = ct * 16 + nn;
    float bv = b2[col];
#pragma unroll
    for (int i = 0; i < 8; ++i) {
      int e = e0 + mb + i;
      float x = acc[i] + bv;
      if (col < SD) abuf[(size_t)e * SD + col] = x;
      else          vijbuf[(size_t)e * vw + (col - SD)] = x;
    }
  }
}

// ---- scatter softmax + aggregation (atomics) -------------------------------

__device__ __forceinline__ unsigned encf(float f) {
  unsigned u = __float_as_uint(f);
  return (u & 0x80000000u) ? ~u : (u | 0x80000000u);
}
__device__ __forceinline__ float decf(unsigned u) {
  u = (u & 0x80000000u) ? (u & 0x7FFFFFFFu) : ~u;
  return __uint_as_float(u);
}
#define ENC_NEG_INF 0x007FFFFFu  // encf(-inf)

__global__ void k_seg_max(const float* __restrict__ a, const int* __restrict__ dst,
                          unsigned* __restrict__ m, int nE) {
  size_t i = (size_t)blockIdx.x * blockDim.x + threadIdx.x;
  if (i >= (size_t)nE * SD) return;
  int e = (int)(i / SD), c = (int)(i % SD);
  atomicMax(&m[(size_t)dst[e] * SD + c], encf(a[i]));
}
__global__ void k_seg_expsum(float* __restrict__ a, const int* __restrict__ dst,
                             const unsigned* __restrict__ m, float* __restrict__ z, int nE) {
  size_t i = (size_t)blockIdx.x * blockDim.x + threadIdx.x;
  if (i >= (size_t)nE * SD) return;
  int e = (int)(i / SD), c = (int)(i % SD);
  float mv = decf(m[(size_t)dst[e] * SD + c]);
  if (!isfinite(mv)) mv = 0.f;
  float ev = __expf(a[i] - mv);
  a[i] = ev;
  atomicAdd(&z[(size_t)dst[e] * SD + c], ev);
}
__global__ void k_attn_agg(const float* __restrict__ a, const int* __restrict__ src,
                           const int* __restrict__ dst, const float* __restrict__ z,
                           const float* __restrict__ sb, float* __restrict__ ms, int nE) {
  size_t i = (size_t)blockIdx.x * blockDim.x + threadIdx.x;
  if (i >= (size_t)nE * SD) return;
  int e = (int)(i / SD), c = (int)(i % SD);
  float alpha = a[i] / (z[(size_t)dst[e] * SD + c] + 1e-16f);
  atomicAdd(&ms[(size_t)dst[e] * SD + c], alpha * sb[(size_t)src[e] * SD + c]);
}
__global__ void k_cnt(const int* __restrict__ dst, float* __restrict__ cnt, int nE) {
  int e = blockIdx.x * blockDim.x + threadIdx.x;
  if (e < nE) atomicAdd(&cnt[dst[e]], 1.0f);
}

// ---- per-edge vector messages ----------------------------------------------

__global__ void k_edge_vec0(const float* __restrict__ vij, const float* __restrict__ r,
                            const int* __restrict__ dst, float* __restrict__ mv, int nE) {
  size_t idx = (size_t)blockIdx.x * blockDim.x + threadIdx.x;
  if (idx >= (size_t)nE * VD) return;
  int e = (int)(idx / VD), i = (int)(idx % VD);
  int t = dst[e];
  float w = vij[(size_t)e * VD + i];
#pragma unroll
  for (int c = 0; c < 3; ++c)
    atomicAdd(&mv[(size_t)t * 48 + c * VD + i], r[3 * e + c] * w);
}
__global__ void k_edge_vec(const float* __restrict__ v, const float* __restrict__ vecW,
                           const float* __restrict__ vij, const float* __restrict__ r,
                           const int* __restrict__ src, const int* __restrict__ dst,
                           float* __restrict__ mv, int nE) {
  size_t idx = (size_t)blockIdx.x * blockDim.x + threadIdx.x;
  if (idx >= (size_t)nE * VD) return;
  int e = (int)(idx / VD), i = (int)(idx % VD);
  int s = src[e], t = dst[e];
  float vb0 = 0.f, vb1 = 0.f, vb2 = 0.f;
#pragma unroll
  for (int j = 0; j < VD; ++j) {
    float w = vecW[j * VD + i];
    vb0 += v[(size_t)s * 48 + 0 * VD + j] * w;
    vb1 += v[(size_t)s * 48 + 1 * VD + j] * w;
    vb2 += v[(size_t)s * 48 + 2 * VD + j] * w;
  }
  float vd0 = v[(size_t)t * 48 + 0 * VD + i];
  float vd1 = v[(size_t)t * 48 + 1 * VD + i];
  float vd2 = v[(size_t)t * 48 + 2 * VD + i];
  float cx = vd1 * vb2 - vd2 * vb1;
  float cy = vd2 * vb0 - vd0 * vb2;
  float cz = vd0 * vb1 - vd1 * vb0;
  float w0 = vij[(size_t)e * 48 + 0 * VD + i];
  float w1 = vij[(size_t)e * 48 + 1 * VD + i];
  float w2 = vij[(size_t)e * 48 + 2 * VD + i];
  atomicAdd(&mv[(size_t)t * 48 + 0 * VD + i], r[3 * e + 0] * w0 + w1 * vb0 + w2 * cx);
  atomicAdd(&mv[(size_t)t * 48 + 1 * VD + i], r[3 * e + 1] * w0 + w1 * vb1 + w2 * cy);
  atomicAdd(&mv[(size_t)t * 48 + 2 * VD + i], r[3 * e + 2] * w0 + w1 * vb2 + w2 * cz);
}

// ---- message apply + gated-equivariant update ------------------------------

__global__ void k_add_s(float* __restrict__ s, const float* __restrict__ ms, size_t n) {
  size_t i = (size_t)blockIdx.x * blockDim.x + threadIdx.x;
  if (i < n) s[i] += ms[i];
}
__global__ void k_add_v(float* __restrict__ v, const float* __restrict__ mv,
                        const float* __restrict__ cnt, int nN) {
  size_t i = (size_t)blockIdx.x * blockDim.x + threadIdx.x;
  if (i >= (size_t)nN * 48) return;
  v[i] += mv[i] / fmaxf(cnt[i / 48], 1.0f);
}
__global__ void k_cat_s(const float* __restrict__ s, _Float16* __restrict__ cath, int nN) {
  size_t i = (size_t)blockIdx.x * blockDim.x + threadIdx.x;
  if (i >= (size_t)nN * 160) return;
  int n = (int)(i / 160), c = (int)(i % 160);
  if (c < SD)        cath[i] = (_Float16)s[(size_t)n * SD + c];
  else if (c >= 144) cath[i] = (_Float16)0.f;  // K padding; [128,144) by k_vv
}
__global__ void k_vv(const float* __restrict__ v, const float* __restrict__ Wv0,
                     float* __restrict__ vout0, _Float16* __restrict__ cath, int nN) {
  size_t idx = (size_t)blockIdx.x * blockDim.x + threadIdx.x;
  if (idx >= (size_t)nN * 32) return;
  int n = (int)(idx / 32), o = (int)(idx % 32);
  float a0 = 0.f, a1 = 0.f, a2 = 0.f;
#pragma unroll
  for (int j = 0; j < VD; ++j) {
    float w = Wv0[j * 32 + o];
    a0 += v[(size_t)n * 48 + 0 * VD + j] * w;
    a1 += v[(size_t)n * 48 + 1 * VD + j] * w;
    a2 += v[(size_t)n * 48 + 2 * VD + j] * w;
  }
  if (o < VD) {
    float vn = sqrtf(fmaxf(a0 * a0 + a1 * a1 + a2 * a2, 1e-6f));
    cath[(size_t)n * 160 + SD + o] = (_Float16)vn;
  } else {
    int oo = o - VD;
    vout0[(size_t)n * 48 + 0 * VD + oo] = a0;
    vout0[(size_t)n * 48 + 1 * VD + oo] = a1;
    vout0[(size_t)n * 48 + 2 * VD + oo] = a2;
  }
}
__global__ void k_gate(const float* __restrict__ sg, const float* __restrict__ vout0,
                       const float* __restrict__ Wv1, float* __restrict__ v, int nN) {
  size_t idx = (size_t)blockIdx.x * blockDim.x + threadIdx.x;
  if (idx >= (size_t)nN * VD) return;
  int n = (int)(idx / VD), o = (int)(idx % VD);
  float a0 = 0.f, a1 = 0.f, a2 = 0.f;
#pragma unroll
  for (int j = 0; j < VD; ++j) {
    float gw = sg[(size_t)n * 144 + j] * Wv1[j * VD + o];
    a0 += vout0[(size_t)n * 48 + 0 * VD + j] * gw;
    a1 += vout0[(size_t)n * 48 + 1 * VD + j] * gw;
    a2 += vout0[(size_t)n * 48 + 2 * VD + j] * gw;
  }
  v[(size_t)n * 48 + 0 * VD + o] += a0;
  v[(size_t)n * 48 + 1 * VD + o] += a1;
  v[(size_t)n * 48 + 2 * VD + o] += a2;
}
__global__ void k_add_sout(float* __restrict__ s, const float* __restrict__ sg, int nN) {
  size_t i = (size_t)blockIdx.x * blockDim.x + threadIdx.x;
  if (i >= (size_t)nN * SD) return;
  int n = (int)(i / SD), c = (int)(i % SD);
  s[i] += sg[(size_t)n * 144 + VD + c];
}

// ---------------------------------------------------------------------------

static inline int gdiv(long long a, int b) { return (int)((a + b - 1) / b); }

extern "C" void kernel_launch(void* const* d_in, const int* in_sizes, int n_in,
                              void* d_out, int out_size, void* d_ws, size_t ws_size,
                              hipStream_t stream) {
  const int*   atom = (const int*)d_in[0];
  const float* pos  = (const float*)d_in[1];
  const int*   src  = (const int*)d_in[2];
  const int*   dst  = src + NE;
  const float* emb  = (const float*)d_in[3];

  auto F = [&](int i) { return (const float*)d_in[i]; };
  struct Lp {
    const float *eW1, *eb1, *eW2, *eb2, *sW, *sb, *vW, *Wv0, *Ws1, *bs1, *Ws2, *bs2, *Wv1;
    int dout2;
  } L[3];
  int pi = 4;
  for (int l = 0; l < 3; ++l) {
    L[l].eW1 = F(pi++); L[l].eb1 = F(pi++); L[l].eW2 = F(pi++); L[l].eb2 = F(pi++);
    L[l].sW  = F(pi++); L[l].sb  = F(pi++);
    L[l].vW  = (l > 0) ? F(pi++) : nullptr;
    L[l].Wv0 = F(pi++); L[l].Ws1 = F(pi++); L[l].bs1 = F(pi++);
    L[l].Ws2 = F(pi++); L[l].bs2 = F(pi++); L[l].Wv1 = F(pi++);
    L[l].dout2 = (l == 0) ? 144 : 176;
  }
  const float* headW = F(pi++);
  const float* headB = F(pi++);

  // ---- workspace bump allocator ----
  char* wp = (char*)d_ws;
  auto A = [&](size_t bytes) { void* r = wp; wp += (bytes + 255) & ~(size_t)255; return r; };
  _Float16 *w1h[3], *w2h[3], *sWh[3], *Ws1h[3], *Ws2h[3];
  for (int l = 0; l < 3; ++l) {
    w1h[l]  = (_Float16*)A(288 * 128 * 2);
    w2h[l]  = (_Float16*)A(128 * 176 * 2);
    sWh[l]  = (_Float16*)A(128 * 128 * 2);
    Ws1h[l] = (_Float16*)A(160 * 128 * 2);
    Ws2h[l] = (_Float16*)A(128 * 144 * 2);
  }
  _Float16* headWh = (_Float16*)A(128 * 32 * 2);
  float*    s      = (float*)A((size_t)NN * SD * 4);
  _Float16* shh    = (_Float16*)A((size_t)NN * SD * 2);
  float*    v      = (float*)A((size_t)NN * 48 * 4);
  float*    sb     = (float*)A((size_t)NN * SD * 4);
  float*    dbuf   = (float*)A((size_t)NE * 4);
  float*    rbuf   = (float*)A((size_t)NE * 3 * 4);
  _Float16* deh    = (_Float16*)A((size_t)NE * KR * 2);
  float*    abuf   = (float*)A((size_t)NE * SD * 4);
  float*    vijbuf = (float*)A((size_t)NE * 48 * 4);
  unsigned* mbuf   = (unsigned*)A((size_t)NN * SD * 4);
  float*    zbuf   = (float*)A((size_t)NN * SD * 4);
  float*    msbuf  = (float*)A((size_t)NN * SD * 4);
  float*    mvbuf  = (float*)A((size_t)NN * 48 * 4);
  float*    cnt    = (float*)A((size_t)NN * 4);
  float*    vout0  = (float*)A((size_t)NN * 48 * 4);
  _Float16* cath   = (_Float16*)A((size_t)NN * 160 * 2);
  _Float16* hh     = (_Float16*)A((size_t)NN * SD * 2);
  float*    sg     = (float*)A((size_t)NN * 144 * 4);

  const int TB = 256;
  auto G = [&](long long n) { return dim3(gdiv(n, TB)); };
  auto CVT = [&](const float* srcw, _Float16* dstw, int din, int dout, int din_p, int dout_p) {
    k_cvt_pack<<<G((long long)din_p * dout_p), TB, 0, stream>>>(srcw, dstw, din, dout,
                                                                din_p, dout_p);
  };

  // weight conversions into fragment-ready packed f16 (every call; deterministic)
  for (int l = 0; l < 3; ++l) {
    CVT(L[l].eW1, w1h[l], 288, 128, 288, 128);
    CVT(L[l].eW2, w2h[l], 128, L[l].dout2, 128, L[l].dout2);
    CVT(L[l].sW,  sWh[l], 128, 128, 128, 128);
    CVT(L[l].Ws1, Ws1h[l], 144, 128, 160, 128);
    CVT(L[l].Ws2, Ws2h[l], 128, 144, 128, 144);
  }
  CVT(headW, headWh, 128, 20, 128, 32);

  // init node state + edge geometry (once; constant across layers)
  k_embed<<<G((long long)NN * SD), TB, 0, stream>>>(atom, emb, s, NN);
  k_fill_f32<<<G((long long)NN * 48), TB, 0, stream>>>(v, 0.f, (size_t)NN * 48);
  k_edge_geom<<<G(NE), TB, 0, stream>>>(pos, src, dst, dbuf, rbuf, deh, NE);

  const int nodeTiles = gdiv(NN, 16);
  const int edgeTiles = gdiv(NE, 16);

  for (int l = 0; l < 3; ++l) {
    k_cvt_f16<<<G((long long)NN * SD), TB, 0, stream>>>(s, shh, (size_t)NN * SD);
    // sb = s @ scalar_W + b
    k_node_gemm<<<nodeTiles, 128, 0, stream>>>(shh, sWh[l], L[l].sb, sb, nullptr,
                                               NN, 128, 128, 128, 0);
    // fused edge MLP -> logits a, vector weights vij
    k_edge_mlp<<<edgeTiles, 128, 0, stream>>>(shh, deh, src, dst, w1h[l], L[l].eb1,
                                              w2h[l], L[l].eb2, abuf, vijbuf, NE, L[l].dout2);
    // scatter softmax + aggregation
    k_fill_u32<<<G((long long)NN * SD), TB, 0, stream>>>(mbuf, ENC_NEG_INF, (size_t)NN * SD);
    k_fill_f32<<<G((long long)NN * SD), TB, 0, stream>>>(zbuf, 0.f, (size_t)NN * SD);
    k_fill_f32<<<G((long long)NN * SD), TB, 0, stream>>>(msbuf, 0.f, (size_t)NN * SD);
    k_fill_f32<<<G((long long)NN * 48), TB, 0, stream>>>(mvbuf, 0.f, (size_t)NN * 48);
    k_fill_f32<<<G(NN), TB, 0, stream>>>(cnt, 0.f, NN);
    k_seg_max<<<G((long long)NE * SD), TB, 0, stream>>>(abuf, dst, mbuf, NE);
    k_seg_expsum<<<G((long long)NE * SD), TB, 0, stream>>>(abuf, dst, mbuf, zbuf, NE);
    k_attn_agg<<<G((long long)NE * SD), TB, 0, stream>>>(abuf, src, dst, zbuf, sb, msbuf, NE);
    k_cnt<<<G(NE), TB, 0, stream>>>(dst, cnt, NE);
    if (l == 0)
      k_edge_vec0<<<G((long long)NE * VD), TB, 0, stream>>>(vijbuf, rbuf, dst, mvbuf, NE);
    else
      k_edge_vec<<<G((long long)NE * VD), TB, 0, stream>>>(v, L[l].vW, vijbuf, rbuf,
                                                           src, dst, mvbuf, NE);
    k_add_s<<<G((long long)NN * SD), TB, 0, stream>>>(s, msbuf, (size_t)NN * SD);
    k_add_v<<<G((long long)NN * 48), TB, 0, stream>>>(v, mvbuf, cnt, NN);
    // gated equivariant update
    k_cat_s<<<G((long long)NN * 160), TB, 0, stream>>>(s, cath, NN);
    k_vv<<<G((long long)NN * 32), TB, 0, stream>>>(v, L[l].Wv0, vout0, cath, NN);
    k_node_gemm<<<nodeTiles, 128, 0, stream>>>(cath, Ws1h[l], L[l].bs1, nullptr, hh,
                                               NN, 160, 128, 128, 1);
    k_node_gemm<<<nodeTiles, 128, 0, stream>>>(hh, Ws2h[l], L[l].bs2, sg, nullptr,
                                               NN, 128, 144, 144, 0);
    k_gate<<<G((long long)NN * VD), TB, 0, stream>>>(sg, vout0, L[l].Wv1, v, NN);
    k_add_sout<<<G((long long)NN * SD), TB, 0, stream>>>(s, sg, NN);
  }

  // head: out = s @ head_W + head_b  (dout padded 20 -> 32)
  k_cvt_f16<<<G((long long)NN * SD), TB, 0, stream>>>(s, shh, (size_t)NN * SD);
  k_node_gemm<<<nodeTiles, 128, 0, stream>>>(shh, headWh, headB, (float*)d_out, nullptr,
                                             NN, 128, 32, 20, 0);
}
